// additive_attention_model_70695161692409
// MI455X (gfx1250) — matrix operand order repaired and checked by reading, not compiled, not run on hardware
//
#include <hip/hip_runtime.h>
#include <hip/hip_bf16.h>

typedef _Float16 f16;
typedef __attribute__((ext_vector_type(16))) _Float16 v16h;
typedef __attribute__((ext_vector_type(8)))  _Float16 v8h;
typedef __attribute__((ext_vector_type(8)))  float    v8f;
typedef __attribute__((ext_vector_type(4))) unsigned int u32x4;
typedef __attribute__((ext_vector_type(8))) int          i32x8;
typedef __attribute__((ext_vector_type(4))) int          i32x4;

#define HAS_TDM (__has_builtin(__builtin_amdgcn_tensor_load_to_lds) && \
                 __has_builtin(__builtin_amdgcn_s_wait_tensorcnt))

// ---------------------------------------------------------------------------
// WMMA helpers (CDNA5 16x16x32 f16 -> f32), layouts per CDNA5 ISA
// ---------------------------------------------------------------------------
__device__ inline v8f wmma_f32(v16h a, v16h b, v8f c) {
  return __builtin_amdgcn_wmma_f32_16x16x32_f16(false, a, false, b, (short)0, c,
                                                false, false);
}

// A tile 16x32 (MxK) from row-major source (global or LDS), vectorized.
__device__ inline v16h wmma_load_a(const f16* __restrict__ A, int lda, int m0, int k0) {
  const int lane = threadIdx.x & 31;
  const int m = m0 + (lane & 15);
  const int g = lane >> 4;
  const f16* p = A + (size_t)m * lda + k0 + 8 * g;
  v8h lo = *(const v8h*)p;
  v8h hi = *(const v8h*)(p + 16);
  return __builtin_shufflevector(lo, hi, 0, 1, 2, 3, 4, 5, 6, 7,
                                 8, 9, 10, 11, 12, 13, 14, 15);
}

// B fragments from pre-packed tile-major storage:
//   Bp[((k>>5)*nbTot + (n>>4))*512 + lane*16 + elem],
//   lane = ((k>>3)&1)*16 + (n&15), elem = (k&7) + ((k>>4)&1)*8  => one 32B load per lane.
__device__ inline v16h wmma_load_b_packed(const f16* __restrict__ Bp, int nbTot,
                                          int k0, int n0) {
  const int lane = threadIdx.x & 31;
  const f16* p = Bp + ((size_t)((k0 >> 5) * nbTot + (n0 >> 4)) * 32 + lane) * 16;
  return *(const v16h*)p;
}

// Same fragment shape from an LDS-staged panel (tile index tt within panel).
__device__ inline v16h wmma_load_b_lds(const f16* __restrict__ bs, int tt) {
  const int lane = threadIdx.x & 31;
  return *(const v16h*)(bs + ((size_t)tt * 32 + lane) * 16);
}

// ---------------------------------------------------------------------------
// Tensor Data Mover: 1-row tile copy of `nel` 2-byte elements, global -> LDS.
// D# built per CDNA5 ISA (count=1, data_size=2B,
// tile = nel x 1, tensor_dim0 = nel, type=2). Groups 2/3 unused (<=2D tensor).
// This toolchain exposes the 6-arg builtin (g0,g1,g2,g3,g4,cpol).
// ---------------------------------------------------------------------------
#if HAS_TDM
__device__ inline void tdm_copy_2b(const void* gsrc, void* ldst, unsigned nel) {
  unsigned long long ga = (unsigned long long)gsrc;
  unsigned la = (unsigned)(unsigned long long)
      (__attribute__((address_space(3))) char*)ldst;   // LDS byte offset
  u32x4 g0;
  g0[0] = 1u;                                         // count=1 (user D#)
  g0[1] = la;                                         // lds_addr   [63:32]
  g0[2] = (unsigned)(ga & 0xffffffffu);               // global lo  [95:64]
  g0[3] = (unsigned)((ga >> 32) & 0x01ffffffu)        // global hi  [120:96]
          | (2u << 30);                               // type=2     [127:126]
  i32x8 g1;
  g1[0] = (int)(1u << 16);                            // data_size=1 (2 bytes)
  g1[1] = (int)((nel & 0xffffu) << 16);               // tensor_dim0 lo16 [63:48]
  g1[2] = (int)(((nel >> 16) & 0xffffu) | (1u << 16));// dim0 hi16 | tensor_dim1=1
  g1[3] = (int)((nel & 0xffffu) << 16);               // tile_dim0 [127:112]
  g1[4] = 1;                                          // tile_dim1=1, tile_dim2=0
  g1[5] = (int)nel;                                   // tensor_dim0_stride lo32
  g1[6] = 0;                                          // stride hi | dim1_stride lo
  g1[7] = 0;
  i32x4 z4 = {0, 0, 0, 0};
  i32x8 z8 = {0, 0, 0, 0, 0, 0, 0, 0};
  __builtin_amdgcn_tensor_load_to_lds(g0, g1, z4, z4, z8, 0);
}
#endif

// ---------------------------------------------------------------------------
// Weight repack: f32 B[K,N] (or transpose) -> f16 packed WMMA tiles.
// ---------------------------------------------------------------------------
__global__ void pack_b_f16(const float* __restrict__ src, f16* __restrict__ dst,
                           int K, int N, int Npad, int srcStride, int transSrc) {
  int i = blockIdx.x * 256 + threadIdx.x;
  int total = K * Npad;
  if (i >= total) return;
  int elem = i & 15;
  int lane = (i >> 4) & 31;
  int tile = i >> 9;
  int nbTot = Npad >> 4;
  int kblk = tile / nbTot, nblk = tile - kblk * nbTot;
  int k = (kblk << 5) + ((lane >> 4) << 3) + (elem & 7) + ((elem >> 3) << 4);
  int n = (nblk << 4) + (lane & 15);
  float v = 0.f;
  if (n < N)
    v = transSrc ? src[(size_t)n * srcStride + k] : src[(size_t)k * srcStride + n];
  dst[i] = (f16)v;
}

// tokens [B=32,S=128] -> X f16 [S,B,D=256]
__global__ void embed_gather(const int* __restrict__ tok, const float* __restrict__ emb,
                             f16* __restrict__ X) {
  int sb = blockIdx.x;            // s*32 + b
  int s = sb >> 5, b = sb & 31;
  int t = tok[b * 128 + s];
  X[(size_t)sb * 256 + threadIdx.x] = (f16)emb[(size_t)t * 256 + threadIdx.x];
}

// ---------------------------------------------------------------------------
// WMMA GEMM with TDM (or fallback) double-buffered LDS staging of packed B.
// Block = 256 threads = 8 waves (2x4), wave tile 32x32, block tile 64x128.
// remapBS: row m = s*32+b  ->  out row b*128+s.
// ---------------------------------------------------------------------------
__global__ __launch_bounds__(256) void gemm_f16_f32(
    const f16* __restrict__ A, const f16* __restrict__ Bp,
    const float* __restrict__ bias, float* __restrict__ C,
    int M, int N, int K, int lda, int nbTot, int ldc, int remapBS)
{
  __shared__ f16 bstage[2][4096];   // 2 x 8KB panels
  const int wave = threadIdx.x >> 5, lane = threadIdx.x & 31;
  const int wm = blockIdx.y * 64 + (wave >> 2) * 32;
  const int bn = blockIdx.x * 128;
  const int wn = bn + (wave & 3) * 32;
  const int nb0 = bn >> 4;
  const int tt0 = (wave & 3) * 2;
  const int nk = K >> 5;

  auto stage = [&](int kc, int buf) {
    const f16* src = Bp + (size_t)(kc * nbTot + nb0) * 512;
#if HAS_TDM
    if (threadIdx.x == 0)
      tdm_copy_2b(src, &bstage[buf][0], 4096u);       // one 8KB DMA per panel
#else
    const f16* s = src + threadIdx.x * 16;
    f16* dst = &bstage[buf][threadIdx.x * 16];
    *(v8h*)dst = *(const v8h*)s;
    *(v8h*)(dst + 8) = *(const v8h*)(s + 8);
#endif
  };

  stage(0, 0);
  v8f acc[2][2] = {};
  for (int kc = 0; kc < nk; ++kc) {
    __syncthreads();                      // prior reads done before next overwrite
    if (kc + 1 < nk) {
      stage(kc + 1, (kc + 1) & 1);
      __builtin_prefetch(Bp + (size_t)((kc + 2) * nbTot + nb0) * 512, 0, 1);
#if HAS_TDM
      if (wave == 0) __builtin_amdgcn_s_wait_tensorcnt(1);  // panel kc landed
#endif
    } else {
#if HAS_TDM
      if (wave == 0) __builtin_amdgcn_s_wait_tensorcnt(0);
#endif
    }
    __syncthreads();                      // publish staged panel to all waves
    const int k0 = kc << 5;
    v16h a0 = wmma_load_a(A, lda, wm, k0);
    v16h a1 = wmma_load_a(A, lda, wm + 16, k0);
    const f16* bs = bstage[kc & 1];
    v16h b0 = wmma_load_b_lds(bs, tt0);
    v16h b1 = wmma_load_b_lds(bs, tt0 + 1);
    acc[0][0] = wmma_f32(a0, b0, acc[0][0]);
    acc[0][1] = wmma_f32(a0, b1, acc[0][1]);
    acc[1][0] = wmma_f32(a1, b0, acc[1][0]);
    acc[1][1] = wmma_f32(a1, b1, acc[1][1]);
  }
  const int nl = lane & 15, mo = (lane >> 4) * 8;
#pragma unroll
  for (int ti = 0; ti < 2; ++ti)
#pragma unroll
    for (int tj = 0; tj < 2; ++tj)
#pragma unroll
      for (int r = 0; r < 8; ++r) {
        int m = wm + 16 * ti + mo + r;
        int n = wn + 16 * tj + nl;
        if (m < M && n < N) {
          float v = acc[ti][tj][r];
          if (bias) v += bias[n];
          int row = m;
          if (remapBS) { int b = m & 31; int s = m >> 5; row = b * 128 + s; }
          C[(size_t)row * ldc + n] = v;
        }
      }
}

// ---------------------------------------------------------------------------
// Encoder GRU scan. grid=2 (dir 0=fwd, 1=bwd), block=512 (16 waves).
// ---------------------------------------------------------------------------
__global__ __launch_bounds__(512) void gru_scan_kernel(
    const float* __restrict__ GXf, const float* __restrict__ GXb,
    const f16* __restrict__ Whf, const f16* __restrict__ Whb,   // packed, nbTot=96
    const float* __restrict__ brf, const float* __restrict__ brb,
    f16* __restrict__ enc, float* __restrict__ h0dec, float* __restrict__ GsAll)
{
  __shared__ f16 hsh[32 * 512];
  const int dir = blockIdx.x;
  const float* GX = dir ? GXb : GXf;
  const f16*  Wh  = dir ? Whb : Whf;
  const float* br = dir ? brb : brf;
  float* Gs = GsAll + (size_t)dir * 32 * 1536;

  for (int i = threadIdx.x; i < 32 * 512; i += blockDim.x) hsh[i] = (f16)0.f;
  __syncthreads();

  const int wave = threadIdx.x >> 5, lane = threadIdx.x & 31;
  const int nl = lane & 15, mo = (lane >> 4) * 8;

  for (int step = 0; step < 128; ++step) {
    const int sin = dir ? (127 - step) : step;
    for (int jj = 0; jj < 2; ++jj) {
      const int job = wave + jj * 16;
      const int mt = (job >> 4) * 16;
      const int c0 = (job & 15) * 32;
      v8f az[2] = {}, ar[2] = {}, ah[2] = {};
      for (int k0 = 0; k0 < 512; k0 += 32) {
        v16h a = wmma_load_a(hsh, 512, mt, k0);
#pragma unroll
        for (int t = 0; t < 2; ++t) {
          az[t] = wmma_f32(a, wmma_load_b_packed(Wh, 96, k0,        c0 + 16 * t), az[t]);
          ar[t] = wmma_f32(a, wmma_load_b_packed(Wh, 96, k0,  512 + c0 + 16 * t), ar[t]);
          ah[t] = wmma_f32(a, wmma_load_b_packed(Wh, 96, k0, 1024 + c0 + 16 * t), ah[t]);
        }
      }
#pragma unroll
      for (int t = 0; t < 2; ++t) {
        const int c = c0 + 16 * t + nl;
#pragma unroll
        for (int r = 0; r < 8; ++r) {
          const int m = mt + mo + r;
          Gs[(size_t)m * 1536 +        c] = az[t][r];
          Gs[(size_t)m * 1536 +  512 + c] = ar[t][r];
          Gs[(size_t)m * 1536 + 1024 + c] = ah[t][r];
        }
      }
    }
    __threadfence();
    __syncthreads();
    const float* gx0 = GX + (size_t)sin * 32 * 1536;
    for (int idx = threadIdx.x; idx < 32 * 512; idx += blockDim.x) {
      const int m = idx >> 9, c = idx & 511;
      const float* gx = gx0 + (size_t)m * 1536;
      const float* g  = Gs + (size_t)m * 1536;
      float z  = 1.f / (1.f + __expf(-(gx[c]       + g[c]       + br[c])));
      float rr = 1.f / (1.f + __expf(-(gx[512 + c] + g[512 + c] + br[512 + c])));
      float hc = tanhf(gx[1024 + c] + rr * (g[1024 + c] + br[1024 + c]));
      float hold = (float)hsh[idx];
      float hn = z * hold + (1.f - z) * hc;
      hsh[idx] = (f16)hn;
      enc[((size_t)sin * 32 + m) * 1024 + dir * 512 + c] = (f16)hn;
    }
    __syncthreads();
  }
  if (dir == 1)
    for (int i = threadIdx.x; i < 32 * 512; i += blockDim.x)
      h0dec[i] = (float)hsh[i];
}

// ---------------------------------------------------------------------------
// Attention + decoder GRU scan. 1 block, 512 threads (16 waves).
// ---------------------------------------------------------------------------
__global__ __launch_bounds__(512) void decoder_kernel(
    const f16* __restrict__ enc,       // [S,B,2H]
    const float* __restrict__ Uenc,    // [S,B,A]
    const f16* __restrict__ WaTp,      // packed [512,128], nbTot=8
    const float* __restrict__ va,      // [A]
    const f16* __restrict__ Wxdp,      // packed [2048,1536], nbTot=96
    const f16* __restrict__ Whdp,      // packed [512,1536],  nbTot=96
    const float* __restrict__ bid, const float* __restrict__ brd,
    const float* __restrict__ h0,      // [32][512] f32
    f16* __restrict__ din,             // [32][2048] scratch (global)
    float* __restrict__ Gs,            // [32][2048] scratch (global)
    f16* __restrict__ dect)            // [S,B,H]
{
  __shared__ f16  hsh[32 * 512];   // 32 KB
  __shared__ f16  whs[32 * 128];   //  8 KB
  __shared__ float scs[32 * 128];  // 16 KB
  for (int i = threadIdx.x; i < 32 * 512; i += blockDim.x) hsh[i] = (f16)h0[i];
  __syncthreads();

  const int wave = threadIdx.x >> 5, lane = threadIdx.x & 31;
  const int nl = lane & 15, mo = (lane >> 4) * 8;

  for (int step = 0; step < 128; ++step) {
    // 1) wh = h @ W_a^T  [32x128]; 16 tiles, one per wave
    {
      const int mt = (wave >> 3) * 16, nt = (wave & 7) * 16;
      v8f acc = {};
      for (int k0 = 0; k0 < 512; k0 += 32)
        acc = wmma_f32(wmma_load_a(hsh, 512, mt, k0),
                       wmma_load_b_packed(WaTp, 8, k0, nt), acc);
#pragma unroll
      for (int r = 0; r < 8; ++r)
        whs[(mt + mo + r) * 128 + nt + nl] = (f16)acc[r];
    }
    __syncthreads();
    // 2) score[b][s] = sum_a tanh(wh[b][a] + Uenc[s][b][a]) * v[a]
    for (int idx = threadIdx.x; idx < 32 * 128; idx += blockDim.x) {
      const int b = idx >> 7, s = idx & 127;
      const float* ue = Uenc + ((size_t)s * 32 + b) * 128;
      const f16* wh = whs + b * 128;
      float acc = 0.f;
      for (int a = 0; a < 128; ++a)
        acc += tanhf((float)wh[a] + ue[a]) * va[a];
      scs[idx] = acc;
    }
    __syncthreads();
    // 3) softmax over s, one wave per row (rows w and w+16)
    for (int jj = 0; jj < 2; ++jj) {
      const int b = wave + jj * 16;
      float v0[4], mx = -3.4e38f;
#pragma unroll
      for (int i = 0; i < 4; ++i) { v0[i] = scs[b * 128 + lane + 32 * i]; mx = fmaxf(mx, v0[i]); }
#pragma unroll
      for (int msk = 16; msk >= 1; msk >>= 1) mx = fmaxf(mx, __shfl_xor(mx, msk, 32));
      float sum = 0.f;
#pragma unroll
      for (int i = 0; i < 4; ++i) { v0[i] = __expf(v0[i] - mx); sum += v0[i]; }
#pragma unroll
      for (int msk = 16; msk >= 1; msk >>= 1) sum += __shfl_xor(sum, msk, 32);
      const float inv = 1.f / sum;
#pragma unroll
      for (int i = 0; i < 4; ++i) scs[b * 128 + lane + 32 * i] = v0[i] * inv;
    }
    __syncthreads();
    // 4) din = [enc_t[step], ctx]
    for (int i = threadIdx.x; i < 32 * 1024; i += blockDim.x) {
      const int b = i >> 10, j = i & 1023;
      din[b * 2048 + j] = enc[((size_t)step * 32 + b) * 1024 + j];
      float acc = 0.f;
      const float* al = scs + b * 128;
      for (int s = 0; s < 128; ++s)
        acc += al[s] * (float)enc[((size_t)s * 32 + b) * 1024 + j];
      din[b * 2048 + 1024 + j] = (f16)acc;
    }
    __threadfence();
    __syncthreads();
    // 5) gates
    for (int jj = 0; jj < 2; ++jj) {
      const int job = wave + jj * 16;
      const int mt = (job >> 4) * 16, c0 = (job & 15) * 32;
      v8f az[2] = {}, ar[2] = {}, axh[2] = {}, ahh[2] = {};
      for (int k0 = 0; k0 < 2048; k0 += 32) {
        v16h a = wmma_load_a(din, 2048, mt, k0);
#pragma unroll
        for (int t = 0; t < 2; ++t) {
          az[t]  = wmma_f32(a, wmma_load_b_packed(Wxdp, 96, k0,        c0 + 16 * t), az[t]);
          ar[t]  = wmma_f32(a, wmma_load_b_packed(Wxdp, 96, k0,  512 + c0 + 16 * t), ar[t]);
          axh[t] = wmma_f32(a, wmma_load_b_packed(Wxdp, 96, k0, 1024 + c0 + 16 * t), axh[t]);
        }
      }
      for (int k0 = 0; k0 < 512; k0 += 32) {
        v16h a = wmma_load_a(hsh, 512, mt, k0);
#pragma unroll
        for (int t = 0; t < 2; ++t) {
          az[t]  = wmma_f32(a, wmma_load_b_packed(Whdp, 96, k0,        c0 + 16 * t), az[t]);
          ar[t]  = wmma_f32(a, wmma_load_b_packed(Whdp, 96, k0,  512 + c0 + 16 * t), ar[t]);
          ahh[t] = wmma_f32(a, wmma_load_b_packed(Whdp, 96, k0, 1024 + c0 + 16 * t), ahh[t]);
        }
      }
#pragma unroll
      for (int t = 0; t < 2; ++t) {
        const int c = c0 + 16 * t + nl;
#pragma unroll
        for (int r = 0; r < 8; ++r) {
          const int m = mt + mo + r;
          Gs[(size_t)m * 2048 +         c] = az[t][r];
          Gs[(size_t)m * 2048 +  512 +  c] = ar[t][r];
          Gs[(size_t)m * 2048 + 1024 +  c] = axh[t][r];
          Gs[(size_t)m * 2048 + 1536 +  c] = ahh[t][r];
        }
      }
    }
    __threadfence();
    __syncthreads();
    // 6) gate math + h update + dec output
    for (int idx = threadIdx.x; idx < 32 * 512; idx += blockDim.x) {
      const int m = idx >> 9, c = idx & 511;
      const float* g = Gs + (size_t)m * 2048;
      float z  = 1.f / (1.f + __expf(-(g[c]       + bid[c]       + brd[c])));
      float rr = 1.f / (1.f + __expf(-(g[512 + c] + bid[512 + c] + brd[512 + c])));
      float hc = tanhf(g[1024 + c] + bid[1024 + c] + rr * (g[1536 + c] + brd[1024 + c]));
      float hold = (float)hsh[idx];
      float hn = z * hold + (1.f - z) * hc;
      hsh[idx] = (f16)hn;
      dect[((size_t)step * 32 + m) * 512 + c] = (f16)hn;
    }
    __syncthreads();
  }
}

// ---------------------------------------------------------------------------
// Row softmax over d_out, one block (256 threads) per row of length N
// ---------------------------------------------------------------------------
__global__ __launch_bounds__(256) void softmax_rows(float* __restrict__ out, int N) {
  __shared__ float red[8];
  float* row = out + (size_t)blockIdx.x * N;
  const int lane = threadIdx.x & 31, wave = threadIdx.x >> 5;
  float mx = -3.4e38f;
  for (int i = threadIdx.x; i < N; i += 256) mx = fmaxf(mx, row[i]);
  for (int m = 16; m >= 1; m >>= 1) mx = fmaxf(mx, __shfl_xor(mx, m, 32));
  if (lane == 0) red[wave] = mx;
  __syncthreads();
  if (threadIdx.x == 0) {
    float v = red[0];
    for (int i = 1; i < 8; ++i) v = fmaxf(v, red[i]);
    red[0] = v;
  }
  __syncthreads();
  mx = red[0];
  __syncthreads();
  float sum = 0.f;
  for (int i = threadIdx.x; i < N; i += 256) {
    float e = __expf(row[i] - mx);
    row[i] = e;
    sum += e;
  }
  for (int m = 16; m >= 1; m >>= 1) sum += __shfl_xor(sum, m, 32);
  if (lane == 0) red[wave] = sum;
  __syncthreads();
  if (threadIdx.x == 0) {
    float v = 0.f;
    for (int i = 0; i < 8; ++i) v += red[i];
    red[0] = v;
  }
  __syncthreads();
  const float inv = 1.f / red[0];
  for (int i = threadIdx.x; i < N; i += 256) row[i] *= inv;
}

// ---------------------------------------------------------------------------
// Host orchestration
// ---------------------------------------------------------------------------
extern "C" void kernel_launch(void* const* d_in, const int* in_sizes, int n_in,
                              void* d_out, int out_size, void* d_ws, size_t ws_size,
                              hipStream_t stream) {
  (void)in_sizes; (void)n_in; (void)out_size; (void)ws_size;
  const int Bc = 32, Sc = 128, Dc = 256, Hc = 512, Ac = 128;
  const int V1 = 32001;          // VOUT+1
  const int NP = 32128;          // padded N (251 * 128) for the logits GEMM

  const int*   tokens = (const int*)d_in[0];
  const float* emb    = (const float*)d_in[1];
  const float* Wx_f   = (const float*)d_in[2];
  const float* Wh_f   = (const float*)d_in[3];
  const float* bi_f   = (const float*)d_in[4];
  const float* br_f   = (const float*)d_in[5];
  const float* Wx_b   = (const float*)d_in[6];
  const float* Wh_b   = (const float*)d_in[7];
  const float* bi_b   = (const float*)d_in[8];
  const float* br_b   = (const float*)d_in[9];
  const float* W_a    = (const float*)d_in[10];
  const float* U_a    = (const float*)d_in[11];
  const float* V_a    = (const float*)d_in[12];
  const float* Wx_d   = (const float*)d_in[13];
  const float* Wh_d   = (const float*)d_in[14];
  const float* bi_d   = (const float*)d_in[15];
  const float* br_d   = (const float*)d_in[16];
  const float* Wo     = (const float*)d_in[17];
  const float* bo     = (const float*)d_in[18];
  float* out = (float*)d_out;

  char* base = (char*)d_ws;
  size_t off = 0;
  auto alloc = [&](size_t bytes) -> char* {
    char* p = base + off;
    off = (off + bytes + 255) & ~(size_t)255;
    return p;
  };
  f16*   Xh   = (f16*)  alloc((size_t)Sc * Bc * Dc * 2);       // [S,B,D]
  f16*   Wxfp = (f16*)  alloc((size_t)Dc * 3 * Hc * 2);        // packed
  f16*   Whfp = (f16*)  alloc((size_t)Hc * 3 * Hc * 2);
  f16*   Wxbp = (f16*)  alloc((size_t)Dc * 3 * Hc * 2);
  f16*   Whbp = (f16*)  alloc((size_t)Hc * 3 * Hc * 2);
  f16*   WaTp = (f16*)  alloc((size_t)Hc * Ac * 2);
  f16*   UaTp = (f16*)  alloc((size_t)2 * Hc * Ac * 2);
  f16*   Wxdp = (f16*)  alloc((size_t)4 * Hc * 3 * Hc * 2);
  f16*   Whdp = (f16*)  alloc((size_t)Hc * 3 * Hc * 2);
  f16*   Wop  = (f16*)  alloc((size_t)Hc * NP * 2);
  float* GXf  = (float*)alloc((size_t)Sc * Bc * 3 * Hc * 4);   // [S,B,3H]
  float* GXb  = (float*)alloc((size_t)Sc * Bc * 3 * Hc * 4);
  f16*   ench = (f16*)  alloc((size_t)Sc * Bc * 2 * Hc * 2);   // [S,B,2H]
  float* Uen  = (float*)alloc((size_t)Sc * Bc * Ac * 4);       // [S,B,A]
  float* h0d  = (float*)alloc((size_t)Bc * Hc * 4);
  f16*   dinb = (f16*)  alloc((size_t)Bc * 4 * Hc * 2);        // [32,2048]
  float* GsE  = (float*)alloc((size_t)2 * Bc * 3 * Hc * 4);
  float* GsD  = (float*)alloc((size_t)Bc * 4 * Hc * 4);
  f16*   dect = (f16*)  alloc((size_t)Sc * Bc * Hc * 2);       // [S,B,H]

  auto cdiv = [](int a, int b) { return (a + b - 1) / b; };

  // weight repack into WMMA fragment order (f32 -> f16)
  pack_b_f16<<<cdiv(Dc*3*Hc,256),256,0,stream>>>(Wx_f, Wxfp, Dc, 3*Hc, 3*Hc, 3*Hc, 0);
  pack_b_f16<<<cdiv(Hc*3*Hc,256),256,0,stream>>>(Wh_f, Whfp, Hc, 3*Hc, 3*Hc, 3*Hc, 0);
  pack_b_f16<<<cdiv(Dc*3*Hc,256),256,0,stream>>>(Wx_b, Wxbp, Dc, 3*Hc, 3*Hc, 3*Hc, 0);
  pack_b_f16<<<cdiv(Hc*3*Hc,256),256,0,stream>>>(Wh_b, Whbp, Hc, 3*Hc, 3*Hc, 3*Hc, 0);
  pack_b_f16<<<cdiv(4*Hc*3*Hc,256),256,0,stream>>>(Wx_d, Wxdp, 4*Hc, 3*Hc, 3*Hc, 3*Hc, 0);
  pack_b_f16<<<cdiv(Hc*3*Hc,256),256,0,stream>>>(Wh_d, Whdp, Hc, 3*Hc, 3*Hc, 3*Hc, 0);
  pack_b_f16<<<cdiv(Hc*NP,256),256,0,stream>>>(Wo, Wop, Hc, V1, NP, V1, 0);
  pack_b_f16<<<cdiv(Hc*Ac,256),256,0,stream>>>(W_a, WaTp, Hc, Ac, Ac, Hc, 1);
  pack_b_f16<<<cdiv(2*Hc*Ac,256),256,0,stream>>>(U_a, UaTp, 2*Hc, Ac, Ac, 2*Hc, 1);

  // embedding gather
  embed_gather<<<Sc*Bc, Dc, 0, stream>>>(tokens, emb, Xh);

  // hoisted input-gate GEMMs (both directions)
  gemm_f16_f32<<<dim3(3*Hc/128, Sc*Bc/64), 256, 0, stream>>>(
      Xh, Wxfp, bi_f, GXf, Sc*Bc, 3*Hc, Dc, Dc, 96, 3*Hc, 0);
  gemm_f16_f32<<<dim3(3*Hc/128, Sc*Bc/64), 256, 0, stream>>>(
      Xh, Wxbp, bi_b, GXb, Sc*Bc, 3*Hc, Dc, Dc, 96, 3*Hc, 0);

  // bidirectional GRU scan
  gru_scan_kernel<<<2, 512, 0, stream>>>(GXf, GXb, Whfp, Whbp, br_f, br_b,
                                         ench, h0d, GsE);

  // hoisted attention projection U_enc = enc @ U_a^T
  gemm_f16_f32<<<dim3(Ac/128, Sc*Bc/64), 256, 0, stream>>>(
      ench, UaTp, nullptr, Uen, Sc*Bc, Ac, 2*Hc, 2*Hc, 8, Ac, 0);

  // attention + decoder GRU scan
  decoder_kernel<<<1, 512, 0, stream>>>(ench, Uen, WaTp, V_a, Wxdp, Whdp,
                                        bi_d, br_d, h0d, dinb, GsD, dect);

  // logits = dec @ Wo + bo, written into d_out ([B,S,V] via remap)
  gemm_f16_f32<<<dim3(cdiv(V1,128), Sc*Bc/64), 256, 0, stream>>>(
      dect, Wop, bo, out, Sc*Bc, V1, Hc, Hc, 2008, V1, 1);

  // in-place softmax over vocab
  softmax_rows<<<Sc*Bc, 256, 0, stream>>>(out, V1);
}
